// RGA_83597243449760
// MI455X (gfx1250) — matrix operand order repaired and checked by reading, not compiled, not run on hardware
//
#include <hip/hip_runtime.h>
#include <hip/hip_bf16.h>
#include <math.h>

// ---------------------------------------------------------------------------
// RGA attention block for MI455X (gfx1250), compile-only target.
//
// Algebraic fusion: the [b,2048,2048] Gs tensor is never materialized.
//   theta/phi [32x2048] = W @ x ;  P1 = gg1 @ phi^T ; P2 = gg2 @ theta^T
//   Gs_joint = relu([P1|P2] @ [theta;phi] + b)   (K=64 instead of K=4096)
// Channel path materializes only Gc [b,256,256].
//
// GEMM contract for WMMA-friendly addressing (all loads 128-bit contiguous):
//   A  : bf16 row-major [M,K]  (lda multiple of 8)
//   Bt : bf16 row-major [N,K]  (B transposed; ldb multiple of 8)
//   K  : multiple of 32 (257-K stages zero-padded to 288)
// Per wave: 16x32 tile, two f32 accumulators sharing one A fragment:
//   A frag : chunks at A[row*lda + k0 + 8*half] and +16    (j<8 / j>=8)
//   B frag : chunks at Bt[col*ldb + k0 + 8*half] and +16    (j<8 / j>=8)
//   C/D    : m = r + 8*half, n = lane%16
// Workspace: ~64 MB.
// Input order: setup_inputs dict order, BN tuples flattened (scale, bias).
// ---------------------------------------------------------------------------

#define NB   8
#define NC   256
#define NHW  2048
#define NIC  32
#define NIS  256
#define NMID 32
#define KP   288   // 257 padded up to multiple of 32

typedef __attribute__((ext_vector_type(16))) __bf16 v16bf;
typedef __attribute__((ext_vector_type(8)))  __bf16 v8bf;
typedef __attribute__((ext_vector_type(8)))  float  v8f;

__device__ __forceinline__ __bf16 us2bf(unsigned short h) {
  union { unsigned short u; __bf16 b; } c; c.u = h; return c.b;
}
__device__ __forceinline__ __bf16 f2bf(float f) {
  union { float f; unsigned u; } c; c.f = f;
  unsigned u = c.u;
  u += 0x7FFFu + ((u >> 16) & 1u);   // round-to-nearest-even
  return us2bf((unsigned short)(u >> 16));
}

// ---------------------------------------------------------------------------
// Batched WMMA GEMM: D[b] = act(A[b] @ B[b] + bias), B given transposed (NxK).
// Optional f32 out D and up to two strided bf16 outs (build concats/transposes
// in the epilogue for free). Block = 128 threads = 4 waves = 32x64 block tile.
// ---------------------------------------------------------------------------
__global__ void wmma_gemm_bf16(
    const __bf16* __restrict__ A, int lda, int aBs,
    const __bf16* __restrict__ Bt, int ldb, int bBs,
    const float* __restrict__ bias,
    float* __restrict__ D, int dMs, int dNs, int dBs,
    __bf16* __restrict__ Dbf, int dbMs, int dbNs, int dbBs,
    __bf16* __restrict__ Dbf2, int d2Ms, int d2Ns, int d2Bs,
    int M, int N, int K, int relu)
{
  const int lane = threadIdx.x & 31;
  const int wave = threadIdx.x >> 5;
  const int m0 = blockIdx.x * 32 + (wave >> 1) * 16;
  const int n0 = blockIdx.y * 64 + (wave & 1) * 32;
  if (m0 >= M || n0 >= N) return;        // wave-uniform -> EXEC stays all-1s
  const int bz = blockIdx.z;
  const int half = lane >> 4;            // 0: lanes 0-15, 1: lanes 16-31
  const int lm   = lane & 15;

  const __bf16* pa  = A  + (size_t)bz * aBs + (size_t)(m0 + lm) * lda + (half << 3);
  const __bf16* pb0 = Bt + (size_t)bz * bBs + (size_t)(n0 + lm) * ldb + (half << 3);
  const __bf16* pb1 = pb0 + (size_t)16 * ldb;

  v8f acc0 = {0.f, 0.f, 0.f, 0.f, 0.f, 0.f, 0.f, 0.f};
  v8f acc1 = {0.f, 0.f, 0.f, 0.f, 0.f, 0.f, 0.f, 0.f};

#pragma unroll 2
  for (int k0 = 0; k0 < K; k0 += 32) {
    v8bf a0  = *(const v8bf*)(pa);       // K = k0 + 8*half + 0..7
    v8bf a1  = *(const v8bf*)(pa + 16);  // K = k0 + 16 + 8*half + 0..7
    v8bf b00 = *(const v8bf*)(pb0);      // K = k0 + 8*half + 0..7
    v8bf b01 = *(const v8bf*)(pb0 + 16); // K = k0 + 16 + 8*half + 0..7
    v8bf b10 = *(const v8bf*)(pb1);
    v8bf b11 = *(const v8bf*)(pb1 + 16);
    v16bf af  = __builtin_shufflevector(a0, a1, 0,1,2,3,4,5,6,7,8,9,10,11,12,13,14,15);
    v16bf bf0 = __builtin_shufflevector(b00, b01, 0,1,2,3,4,5,6,7,8,9,10,11,12,13,14,15);
    v16bf bf1 = __builtin_shufflevector(b10, b11, 0,1,2,3,4,5,6,7,8,9,10,11,12,13,14,15);
    acc0 = __builtin_amdgcn_wmma_f32_16x16x32_bf16(false, af, false, bf0,
                                                   (short)0, acc0, false, false);
    acc1 = __builtin_amdgcn_wmma_f32_16x16x32_bf16(false, af, false, bf1,
                                                   (short)0, acc1, false, false);
    pa += 32; pb0 += 32; pb1 += 32;
  }

  const int nA = n0 + lm, nB = n0 + 16 + lm;
#pragma unroll
  for (int r = 0; r < 8; ++r) {
    const int m = m0 + r + (half << 3);
    float v0 = acc0[r], v1 = acc1[r];
    if (bias) { const float bm = bias[m]; v0 += bm; v1 += bm; }
    if (relu) { v0 = fmaxf(v0, 0.f); v1 = fmaxf(v1, 0.f); }
    if (D) {
      D[(size_t)bz * dBs + (size_t)m * dMs + (size_t)nA * dNs] = v0;
      D[(size_t)bz * dBs + (size_t)m * dMs + (size_t)nB * dNs] = v1;
    }
    if (Dbf) {
      Dbf[(size_t)bz * dbBs + (size_t)m * dbMs + (size_t)nA * dbNs] = f2bf(v0);
      Dbf[(size_t)bz * dbBs + (size_t)m * dbMs + (size_t)nB * dbNs] = f2bf(v1);
    }
    if (Dbf2) {
      Dbf2[(size_t)bz * d2Bs + (size_t)m * d2Ms + (size_t)nA * d2Ns] = f2bf(v0);
      Dbf2[(size_t)bz * d2Bs + (size_t)m * d2Ms + (size_t)nB * d2Ns] = f2bf(v1);
    }
  }
}

// --------------------------- elementwise helpers ---------------------------

__global__ void k_fill_zero_bf16(__bf16* __restrict__ p, int n) {
  int i = blockIdx.x * blockDim.x + threadIdx.x;
  if (i < n) p[i] = us2bf(0);
}

// xT[b,s,c] = bf16(x[b,c,s])   (x transposed for theta/phi GEMM B-operand)
__global__ void k_xT(const float* __restrict__ x, __bf16* __restrict__ xT) {
  int i = blockIdx.x * blockDim.x + threadIdx.x;       // b*HW*C
  if (i >= NB * NHW * NC) return;
  int b = i / (NHW * NC);
  int s = (i / NC) % NHW;
  int c = i % NC;
  xT[i] = f2bf(x[(size_t)b * NC * NHW + (size_t)c * NHW + s]);
}

// out[o,k<Kin] = bf16(w[o,k]*scale[o]); zero-pad k in [Kin,Kout)
__global__ void k_fold_weight(const float* __restrict__ w, const float* __restrict__ s,
                              __bf16* __restrict__ out, int O, int Kin, int Kout) {
  int i = blockIdx.x * blockDim.x + threadIdx.x;
  if (i >= O * Kout) return;
  int o = i / Kout, k = i % Kout;
  out[i] = (k < Kin) ? f2bf(w[(size_t)o * Kin + k] * s[o]) : us2bf(0);
}

// out[k] = (1/O) * sum_o s[o]*w[o,k]  (BN-scale + channel-mean fold of g_x conv)
__global__ void k_fuse_gx(const float* __restrict__ w, const float* __restrict__ s,
                          float* __restrict__ out, int O, int K) {
  int c = blockIdx.x * blockDim.x + threadIdx.x;
  if (c >= K) return;
  float acc = 0.f;
  for (int i = 0; i < O; ++i) acc += s[i] * w[(size_t)i * K + c];
  out[c] = acc / (float)O;
}

// ysT column 0: g_xs[b,s] = wf . x[b,:,s] + mean(bias)
__global__ void k_gxs(const float* __restrict__ x, const float* __restrict__ wf,
                      const float* __restrict__ tb, __bf16* __restrict__ ysT) {
  int i = blockIdx.x * blockDim.x + threadIdx.x;       // b*HW + s
  if (i >= NB * NHW) return;
  int b = i / NHW, s = i % NHW;
  const float* xb = x + (size_t)b * NC * NHW + s;
  float acc = 0.f;
  for (int c = 0; c < NC; ++c) acc += wf[c] * xb[(size_t)c * NHW];
  float bias = 0.f;
  for (int j = 0; j < NIC; ++j) bias += tb[j];
  acc += bias / (float)NIC;
  ysT[(size_t)b * NHW * KP + (size_t)s * KP] = f2bf(acc);
}

// sig_sp[b,s] = sigmoid(s2*(w2 . hmid[b,:,s]) + t2)
__global__ void k_sig_sp(const float* __restrict__ hmid, const float* __restrict__ w2,
                         const float* __restrict__ s2, const float* __restrict__ t2,
                         float* __restrict__ sig) {
  int i = blockIdx.x * blockDim.x + threadIdx.x;
  if (i >= NB * NHW) return;
  int b = i / NHW, s = i % NHW;
  const float* hb = hmid + (size_t)b * NMID * NHW + s;
  float acc = 0.f;
  for (int m = 0; m < NMID; ++m) acc += w2[m] * hb[(size_t)m * NHW];
  float wy = s2[0] * acc + t2[0];
  sig[i] = 1.f / (1.f + expf(-wy));
}

// x2 = sig_sp(b,s) * x; write f32 + bf16 copies ([B,C,HW] = NxK for channel GEMMs)
__global__ void k_apply_sp(const float* __restrict__ x, const float* __restrict__ sig,
                           float* __restrict__ x2f, __bf16* __restrict__ x2bf) {
  int i = blockIdx.x * blockDim.x + threadIdx.x;
  if (i >= NB * NC * NHW) return;
  int b = i / (NC * NHW);
  int s = i % NHW;
  float v = sig[b * NHW + s] * x[i];
  x2f[i] = v;
  x2bf[i] = f2bf(v);
}

// ycT column 0: g_xc[b,d] = wf . x2[b,d,:] + mean(bias)
__global__ void k_gxc(const float* __restrict__ x2f, const float* __restrict__ wf,
                      const float* __restrict__ tb, __bf16* __restrict__ ycT) {
  int i = blockIdx.x * blockDim.x + threadIdx.x;       // b*C + d
  if (i >= NB * NC) return;
  int b = i / NC, d = i % NC;
  const float* xb = x2f + (size_t)b * NC * NHW + (size_t)d * NHW;
  float acc = 0.f;
  for (int s = 0; s < NHW; ++s) acc += wf[s] * xb[s];
  float bias = 0.f;
  for (int j = 0; j < NIS; ++j) bias += tb[j];
  acc += bias / (float)NIS;
  ycT[(size_t)b * NC * KP + (size_t)d * KP] = f2bf(acc);
}

// gcatT[b,d,c] = c<256 ? Gc[b,d,c] : Gc[b,c-256,d]   ([Gc^T;Gc] as NxK operand)
__global__ void k_gcat(const float* __restrict__ gc, __bf16* __restrict__ gcatT) {
  int i = blockIdx.x * blockDim.x + threadIdx.x;       // b*256*512
  if (i >= NB * NC * 2 * NC) return;
  int b = i / (NC * 2 * NC);
  int d = (i / (2 * NC)) % NC;
  int c = i % (2 * NC);
  const float* g = gc + (size_t)b * NC * NC;
  float v = (c < NC) ? g[(size_t)d * NC + c] : g[(size_t)(c - NC) * NC + d];
  gcatT[i] = f2bf(v);
}

// sig_ch[b,d] = sigmoid(s2*(w2 . cmid[b,:,d]) + t2)
__global__ void k_sig_ch(const float* __restrict__ cmid, const float* __restrict__ w2,
                         const float* __restrict__ s2, const float* __restrict__ t2,
                         float* __restrict__ sig) {
  int i = blockIdx.x * blockDim.x + threadIdx.x;
  if (i >= NB * NC) return;
  int b = i / NC, d = i % NC;
  const float* cb = cmid + (size_t)b * NMID * NC + d;
  float acc = 0.f;
  for (int m = 0; m < NMID; ++m) acc += w2[m] * cb[(size_t)m * NC];
  float wy = s2[0] * acc + t2[0];
  sig[i] = 1.f / (1.f + expf(-wy));
}

__global__ void k_apply_ch(const float* __restrict__ x2f, const float* __restrict__ sig,
                           float* __restrict__ out) {
  int i = blockIdx.x * blockDim.x + threadIdx.x;
  if (i >= NB * NC * NHW) return;
  out[i] = sig[i / NHW] * x2f[i];
}

// ---------------------------------------------------------------------------

extern "C" void kernel_launch(void* const* d_in, const int* in_sizes, int n_in,
                              void* d_out, int out_size, void* d_ws, size_t ws_size,
                              hipStream_t stream) {
  (void)in_sizes; (void)n_in; (void)out_size; (void)ws_size;

  const float* x      = (const float*)d_in[0];
  const float* w_thsp = (const float*)d_in[1];
  const float* s_thsp = (const float*)d_in[2];
  const float* t_thsp = (const float*)d_in[3];
  const float* w_phsp = (const float*)d_in[4];
  const float* s_phsp = (const float*)d_in[5];
  const float* t_phsp = (const float*)d_in[6];
  const float* w_gxsp = (const float*)d_in[7];
  const float* s_gxsp = (const float*)d_in[8];
  const float* t_gxsp = (const float*)d_in[9];
  const float* w_ggsp = (const float*)d_in[10];
  const float* s_ggsp = (const float*)d_in[11];
  const float* t_ggsp = (const float*)d_in[12];
  const float* w_sp1  = (const float*)d_in[13];
  const float* s_sp1  = (const float*)d_in[14];
  const float* t_sp1  = (const float*)d_in[15];
  const float* w_sp2  = (const float*)d_in[16];
  const float* s_sp2  = (const float*)d_in[17];
  const float* t_sp2  = (const float*)d_in[18];
  const float* w_thch = (const float*)d_in[19];
  const float* s_thch = (const float*)d_in[20];
  const float* t_thch = (const float*)d_in[21];
  const float* w_phch = (const float*)d_in[22];
  const float* s_phch = (const float*)d_in[23];
  const float* t_phch = (const float*)d_in[24];
  const float* w_gxch = (const float*)d_in[25];
  const float* s_gxch = (const float*)d_in[26];
  const float* t_gxch = (const float*)d_in[27];
  const float* w_ggch = (const float*)d_in[28];
  const float* s_ggch = (const float*)d_in[29];
  const float* t_ggch = (const float*)d_in[30];
  const float* w_ch1  = (const float*)d_in[31];
  const float* s_ch1  = (const float*)d_in[32];
  const float* t_ch1  = (const float*)d_in[33];
  const float* w_ch2  = (const float*)d_in[34];
  const float* s_ch2  = (const float*)d_in[35];
  const float* t_ch2  = (const float*)d_in[36];
  float* out = (float*)d_out;

  // ---- workspace carve-up (256B aligned) ----
  char* ws = (char*)d_ws;
  size_t off = 0;
  auto alloc = [&](size_t bytes) -> void* {
    void* p = ws + off;
    off += (bytes + 255) & ~(size_t)255;
    return p;
  };
  __bf16* xT    = (__bf16*)alloc((size_t)NB * NHW * NC * 2);   // x^T [B,HW,C]
  __bf16* tp    = (__bf16*)alloc((size_t)NB * 64 * NHW * 2);   // [theta;phi] [B,64,HW]
  __bf16* tpT   = (__bf16*)alloc((size_t)NB * NHW * 64 * 2);   // transpose [B,HW,64]
  __bf16* pbuf  = (__bf16*)alloc((size_t)NB * NIS * 64 * 2);   // [P1|P2] [B,256,64]
  __bf16* ysT   = (__bf16*)alloc((size_t)NB * NHW * KP * 2);   // ys^T [B,HW,288] (padded)
  float*  hmid  = (float*)alloc((size_t)NB * NMID * NHW * 4);  // [B,32,HW]
  float*  x2f   = (float*)alloc((size_t)NB * NC * NHW * 4);    // gated x (f32)
  __bf16* x2bf  = (__bf16*)alloc((size_t)NB * NC * NHW * 2);   // gated x (bf16) [B,C,HW]
  __bf16* thcT  = (__bf16*)alloc((size_t)NB * NC * NIS * 2);   // theta_c^T [B,256(d),256(i)]
  __bf16* phcT  = (__bf16*)alloc((size_t)NB * NC * NIS * 2);   // phi_c^T
  float*  gc    = (float*)alloc((size_t)NB * NC * NC * 4);     // Gc [B,256,256]
  __bf16* gcatT = (__bf16*)alloc((size_t)NB * NC * 2 * NC * 2);// [B,256(d),512(c)]
  __bf16* ycT   = (__bf16*)alloc((size_t)NB * NC * KP * 2);    // yc^T [B,256,288] (padded)
  float*  cmid  = (float*)alloc((size_t)NB * NMID * NC * 4);   // [B,32,256]
  float*  sigsp = (float*)alloc((size_t)NB * NHW * 4);
  float*  sigch = (float*)alloc((size_t)NB * NC * 4);
  __bf16* wth   = (__bf16*)alloc((size_t)NIC * NC * 2);
  __bf16* wph   = (__bf16*)alloc((size_t)NIC * NC * 2);
  __bf16* wgg   = (__bf16*)alloc((size_t)NIS * 2 * NHW * 2);   // [256,4096]
  __bf16* wsp1b = (__bf16*)alloc((size_t)NMID * KP * 2);       // [32,288] padded
  __bf16* wthcb = (__bf16*)alloc((size_t)NIS * NHW * 2);
  __bf16* wphcb = (__bf16*)alloc((size_t)NIS * NHW * 2);
  __bf16* wggcb = (__bf16*)alloc((size_t)NIS * 2 * NC * 2);    // [256,512]
  __bf16* wch1b = (__bf16*)alloc((size_t)NMID * KP * 2);       // [32,288] padded
  float*  gxspf = (float*)alloc((size_t)NC * 4);
  float*  gxchf = (float*)alloc((size_t)NHW * 4);

  auto cdiv = [](int a, int b) { return (a + b - 1) / b; };
  auto gemm = [&](const __bf16* A, int lda, int aBs,
                  const __bf16* Bt, int ldb, int bBs,
                  const float* bias,
                  float* D, int dMs, int dNs, int dBs,
                  __bf16* Dbf, int dbMs, int dbNs, int dbBs,
                  __bf16* Dbf2, int d2Ms, int d2Ns, int d2Bs,
                  int M, int N, int K, int relu) {
    dim3 grid(cdiv(M, 32), cdiv(N, 64), NB);
    wmma_gemm_bf16<<<grid, dim3(128), 0, stream>>>(
        A, lda, aBs, Bt, ldb, bBs, bias, D, dMs, dNs, dBs,
        Dbf, dbMs, dbNs, dbBs, Dbf2, d2Ms, d2Ns, d2Bs, M, N, K, relu);
  };

  // ---- prep: zero pads, transposes, BN-scale folding ----
  { int n = NB * NHW * KP; k_fill_zero_bf16<<<cdiv(n, 256), 256, 0, stream>>>(ysT, n); }
  { int n = NB * NC * KP;  k_fill_zero_bf16<<<cdiv(n, 256), 256, 0, stream>>>(ycT, n); }
  { int n = NB * NHW * NC; k_xT<<<cdiv(n, 256), 256, 0, stream>>>(x, xT); }
  k_fold_weight<<<cdiv(NIC * NC, 256), 256, 0, stream>>>(w_thsp, s_thsp, wth, NIC, NC, NC);
  k_fold_weight<<<cdiv(NIC * NC, 256), 256, 0, stream>>>(w_phsp, s_phsp, wph, NIC, NC, NC);
  k_fold_weight<<<cdiv(NIS * 2 * NHW, 256), 256, 0, stream>>>(w_ggsp, s_ggsp, wgg, NIS, 2 * NHW, 2 * NHW);
  k_fold_weight<<<cdiv(NMID * KP, 256), 256, 0, stream>>>(w_sp1, s_sp1, wsp1b, NMID, 257, KP);
  k_fold_weight<<<cdiv(NIS * NHW, 256), 256, 0, stream>>>(w_thch, s_thch, wthcb, NIS, NHW, NHW);
  k_fold_weight<<<cdiv(NIS * NHW, 256), 256, 0, stream>>>(w_phch, s_phch, wphcb, NIS, NHW, NHW);
  k_fold_weight<<<cdiv(NIS * 2 * NC, 256), 256, 0, stream>>>(w_ggch, s_ggch, wggcb, NIS, 2 * NC, 2 * NC);
  k_fold_weight<<<cdiv(NMID * KP, 256), 256, 0, stream>>>(w_ch1, s_ch1, wch1b, NMID, 257, KP);
  k_fuse_gx<<<cdiv(NC, 256), 256, 0, stream>>>(w_gxsp, s_gxsp, gxspf, NIC, NC);
  k_fuse_gx<<<cdiv(NHW, 256), 256, 0, stream>>>(w_gxch, s_gxch, gxchf, NIS, NHW);

  // ---- spatial relation-aware attention ----
  // theta/phi [32x2048] = W @ x : A=[32,256], Bt=xT [2048(s),256(c)]
  // epilogue -> tp rows (i-major) AND tpT (s-major) simultaneously
  gemm(wth, NC, 0, xT, NC, NHW * NC, t_thsp,
       nullptr, 0, 0, 0,
       tp, NHW, 1, 64 * NHW, tpT, 1, 64, NHW * 64,
       NIC, NHW, NC, 0);
  gemm(wph, NC, 0, xT, NC, NHW * NC, t_phsp,
       nullptr, 0, 0, 0,
       tp + 32 * NHW, NHW, 1, 64 * NHW, tpT + 32, 1, 64, NHW * 64,
       NIC, NHW, NC, 0);
  // ys column 0 = g_xs
  k_gxs<<<cdiv(NB * NHW, 256), 256, 0, stream>>>(x, gxspf, t_gxsp, ysT);
  // P1[o,i] = gg1 . phi[i,:] : Bt = phi rows of tp [32(i),2048(t)]
  gemm(wgg, 2 * NHW, 0, tp + 32 * NHW, NHW, 64 * NHW, nullptr,
       nullptr, 0, 0, 0, pbuf, 64, 1, NIS * 64, nullptr, 0, 0, 0,
       NIS, NIC, NHW, 0);
  // P2[o,i] = gg2 . theta[i,:]
  gemm(wgg + NHW, 2 * NHW, 0, tp, NHW, 64 * NHW, nullptr,
       nullptr, 0, 0, 0, pbuf + 32, 64, 1, NIS * 64, nullptr, 0, 0, 0,
       NIS, NIC, NHW, 0);
  // Gs_joint = relu([P1|P2] @ [theta;phi] + b) -> ysT columns 1..256
  gemm(pbuf, 64, NIS * 64, tpT, 64, NHW * 64, t_ggsp,
       nullptr, 0, 0, 0, ysT + 1, 1, KP, NHW * KP, nullptr, 0, 0, 0,
       NIS, NHW, 64, 1);
  // hmid = relu(w_sp1 @ ys + b), K padded 257->288
  gemm(wsp1b, KP, 0, ysT, KP, NHW * KP, t_sp1,
       hmid, NHW, 1, NMID * NHW, nullptr, 0, 0, 0, nullptr, 0, 0, 0,
       NMID, NHW, KP, 1);
  // sigmoid gate, produce x2 (f32 + bf16)
  k_sig_sp<<<cdiv(NB * NHW, 256), 256, 0, stream>>>(hmid, w_sp2, s_sp2, t_sp2, sigsp);
  k_apply_sp<<<cdiv(NB * NC * NHW, 256), 256, 0, stream>>>(x, sigsp, x2f, x2bf);

  // ---- channel relation-aware attention ----
  // theta_c/phi_c [256x256] = W @ x2^T : Bt = x2bf [256(d),2048(s)]
  // epilogue writes transposed (d-major) for the Gc GEMM
  gemm(wthcb, NHW, 0, x2bf, NHW, NC * NHW, t_thch,
       nullptr, 0, 0, 0, thcT, 1, NIS, NC * NIS, nullptr, 0, 0, 0,
       NIS, NC, NHW, 0);
  gemm(wphcb, NHW, 0, x2bf, NHW, NC * NHW, t_phch,
       nullptr, 0, 0, 0, phcT, 1, NIS, NC * NIS, nullptr, 0, 0, 0,
       NIS, NC, NHW, 0);
  // yc column 0 = g_xc
  k_gxc<<<cdiv(NB * NC, 256), 256, 0, stream>>>(x2f, gxchf, t_gxch, ycT);
  // Gc[c,d] = sum_i theta_c[i,c]*phi_c[i,d] : A=thcT [c,i], Bt=phcT [d,i]
  gemm(thcT, NIS, NC * NIS, phcT, NIS, NC * NIS, nullptr,
       gc, NC, 1, NC * NC, nullptr, 0, 0, 0, nullptr, 0, 0, 0,
       NC, NC, NIS, 0);
  // [Gc^T;Gc] as NxK operand
  k_gcat<<<cdiv(NB * NC * 2 * NC, 256), 256, 0, stream>>>(gc, gcatT);
  // Gc_joint = relu(gg_ch @ gcat + b) -> ycT columns 1..256
  gemm(wggcb, 2 * NC, 0, gcatT, 2 * NC, NC * 2 * NC, t_ggch,
       nullptr, 0, 0, 0, ycT + 1, 1, KP, NC * KP, nullptr, 0, 0, 0,
       NIS, NC, 2 * NC, 1);
  // cmid = relu(w_ch1 @ yc + b), K padded 257->288
  gemm(wch1b, KP, 0, ycT, KP, NC * KP, t_ch1,
       cmid, NC, 1, NMID * NC, nullptr, 0, 0, 0, nullptr, 0, 0, 0,
       NMID, NC, KP, 1);
  // channel sigmoid gate, final output
  k_sig_ch<<<cdiv(NB * NC, 256), 256, 0, stream>>>(cmid, w_ch2, s_ch2, t_ch2, sigch);
  k_apply_ch<<<cdiv(NB * NC * NHW, 256), 256, 0, stream>>>(x2f, sigch, out);
}
